// graph_sage_60309930770472
// MI455X (gfx1250) — matrix operand order repaired and checked by reading, not compiled, not run on hardware
//
#include <hip/hip_runtime.h>
#include <hip/hip_bf16.h>

typedef __attribute__((ext_vector_type(16))) _Float16 v16h;
typedef __attribute__((ext_vector_type(8)))  float    v8f;

#define EPS 1e-5f

struct LBSD { const float *W, *b, *g, *be, *mu, *var; };

// ---------------------------------------------------------------------------
// WMMA fragment builders (CDNA5 16x16x32 f16 layouts, wave32)
// A: 16xK tile in LDS, row-major halves, row stride = strideK
//    lane l: m = l&15, hi = l>>4 ; a[j] -> K = kt*32 + 16*hi + j
// B: W stored row-major [NOUT][KIN] halves  (== B-transposed, Bt[n][k]=W[n][k])
//    lane l: n = nt*16 + (l&15), hi = l>>4 ; b[j] -> K = kt*32 + 16*hi + j
// C: c[v] <-> (M = 8*hi + v, N = l&15)
// ---------------------------------------------------------------------------
__device__ inline v16h frag_A(const _Float16* __restrict__ buf, int strideK,
                              int kt, int lane) {
  int m = lane & 15, hi = lane >> 4;
  const _Float16* p = buf + m * strideK + kt * 32 + 16 * hi;
  v16h a;
#pragma unroll
  for (int j = 0; j < 8; ++j) a[j] = p[j];
#pragma unroll
  for (int j = 0; j < 8; ++j) a[8 + j] = p[8 + j];
  return a;
}

__device__ inline v16h frag_B(const _Float16* __restrict__ w, int strideK,
                              int kt, int nt, int lane) {
  int n = nt * 16 + (lane & 15), hi = lane >> 4;
  const _Float16* p = w + n * strideK + kt * 32 + 16 * hi;
  v16h b;
#pragma unroll
  for (int j = 0; j < 16; ++j) b[j] = p[j];
  return b;
}

__device__ inline float silu_f(float x) { return x / (1.0f + __expf(-x)); }

// GEMM: O[16xNOUT] (f16, LDS) = act(A[16xKIN] @ W[NOUTxKIN]^T + bias)
template <int KIN, int NOUT>
__device__ inline void wave_gemm_lds(const _Float16* __restrict__ A,
                                     const _Float16* __restrict__ W,
                                     const float* __restrict__ bias,
                                     _Float16* __restrict__ O,
                                     int lane, bool silu) {
  constexpr int NT = NOUT / 16, KT = KIN / 32;
  int n0 = lane & 15, hi = lane >> 4;
#pragma unroll
  for (int nt = 0; nt < NT; ++nt) {
    v8f c = {};
#pragma unroll
    for (int kt = 0; kt < KT; ++kt) {
      v16h a = frag_A(A, KIN, kt, lane);
      v16h b = frag_B(W, KIN, kt, nt, lane);
      c = __builtin_amdgcn_wmma_f32_16x16x32_f16(false, a, false, b,
                                                 (short)0, c, false, false);
    }
    int n = nt * 16 + n0;
    float bb = bias[n];
#pragma unroll
    for (int v = 0; v < 8; ++v) {
      int m = 8 * hi + v;
      float x = c[v] + bb;
      if (silu) x = silu_f(x);
      O[m * NOUT + n] = (_Float16)x;
    }
  }
}

// GEMM with f32 global-memory epilogue (row/col guarded)
template <int KIN, int NOUT>
__device__ inline void wave_gemm_global(const _Float16* __restrict__ A,
                                        const _Float16* __restrict__ W,
                                        const float* __restrict__ bias,
                                        float* __restrict__ out, int ldo,
                                        int rowsValid, int colsValid,
                                        int lane, bool silu) {
  constexpr int NT = NOUT / 16, KT = KIN / 32;
  int n0 = lane & 15, hi = lane >> 4;
#pragma unroll
  for (int nt = 0; nt < NT; ++nt) {
    v8f c = {};
#pragma unroll
    for (int kt = 0; kt < KT; ++kt) {
      v16h a = frag_A(A, KIN, kt, lane);
      v16h b = frag_B(W, KIN, kt, nt, lane);
      c = __builtin_amdgcn_wmma_f32_16x16x32_f16(false, a, false, b,
                                                 (short)0, c, false, false);
    }
    int n = nt * 16 + n0;
    float bb = bias[n];
#pragma unroll
    for (int v = 0; v < 8; ++v) {
      int m = 8 * hi + v;
      float x = c[v] + bb;
      if (silu) x = silu_f(x);
      if (m < rowsValid && n < colsValid)
        out[(long long)m * ldo + n] = x;
    }
  }
}

// Stage Linear+BN(eval) folded weights into LDS as f16: Weff = g*rsqrt(var+eps)*W
__device__ inline void stage_lbsd(const LBSD L, _Float16* sW, float* sB,
                                  int dout, int din, int tid, int nthreads) {
  for (int o = tid; o < dout; o += nthreads) {
    float s = L.g[o] * rsqrtf(L.var[o] + EPS);
    sB[o] = (L.b[o] - L.mu[o]) * s + L.be[o];
  }
  for (int t = tid; t < dout * din; t += nthreads) {
    int o = t / din;
    float s = L.g[o] * rsqrtf(L.var[o] + EPS);
    sW[t] = (_Float16)(L.W[t] * s);
  }
}

// ---------------------------------------------------------------------------
// Kernel 1: fused encoder  x[N,128] -> h3[N,32]   (3x LBSD via WMMA)
// ---------------------------------------------------------------------------
__global__ __launch_bounds__(128) void encoder_kernel(
    const float* __restrict__ x, LBSD e0, LBSD e1, LBSD e2,
    float* __restrict__ h3, int nN) {
  __shared__ _Float16 sW0[64 * 128];
  __shared__ _Float16 sW1[64 * 64];
  __shared__ _Float16 sW2[32 * 64];
  __shared__ float sB0[64], sB1[64], sB2[32];
  __shared__ _Float16 sAct0[4][16 * 128];
  __shared__ _Float16 sAct1[4][16 * 64];

  int tid = threadIdx.x, wave = tid >> 5, lane = tid & 31;
  stage_lbsd(e0, sW0, sB0, 64, 128, tid, 128);
  stage_lbsd(e1, sW1, sB1, 64, 64, tid, 128);
  stage_lbsd(e2, sW2, sB2, 32, 64, tid, 128);
  __syncthreads();

  int base = blockIdx.x * 64 + wave * 16;
  // stage 16x128 input tile as f16 (uniform trip count: 2048/32)
  for (int t = lane; t < 16 * 128; t += 32) {
    int r = t >> 7, k = t & 127;
    int gm = base + r;
    int gmc = (gm < nN) ? gm : (nN - 1);
    float v = x[(long long)gmc * 128 + k];
    sAct0[wave][t] = (_Float16)((gm < nN) ? v : 0.0f);
  }
  __syncthreads();

  wave_gemm_lds<128, 64>(sAct0[wave], sW0, sB0, sAct1[wave], lane, true);
  __syncthreads();
  wave_gemm_lds<64, 64>(sAct1[wave], sW1, sB1, sAct0[wave], lane, true);
  __syncthreads();
  int rows = nN - base; rows = rows < 0 ? 0 : (rows > 16 ? 16 : rows);
  wave_gemm_global<64, 32>(sAct0[wave], sW2, sB2,
                           h3 + (long long)base * 32, 32, rows, 32, lane, true);
}

// ---------------------------------------------------------------------------
// Edge kernels: in-degree + scatter-add (mean aggregation numerator)
// ---------------------------------------------------------------------------
__global__ void zero_kernel(float* __restrict__ p, size_t n) {
  size_t i = (size_t)blockIdx.x * blockDim.x + threadIdx.x;
  size_t st = (size_t)gridDim.x * blockDim.x;
  for (; i < n; i += st) p[i] = 0.0f;
}

__global__ void degree_kernel(const int* __restrict__ dst,
                              float* __restrict__ cnt, int nE) {
  int e = blockIdx.x * blockDim.x + threadIdx.x;
  if (e < nE) atomicAdd(&cnt[dst[e]], 1.0f);
}

template <int D>
__global__ void scatter_add(const int* __restrict__ src,
                            const int* __restrict__ dst,
                            const float* __restrict__ h,
                            float* __restrict__ agg, int nE) {
  long long i = (long long)blockIdx.x * blockDim.x + threadIdx.x;
  long long tot = (long long)nE * D;
  if (i >= tot) return;
  int e = (int)(i / D), d = (int)(i % D);
  atomicAdd(&agg[(long long)dst[e] * D + d], h[(long long)src[e] * D + d]);
}

// ---------------------------------------------------------------------------
// Kernel 2: sage_in  h4 = [mean1|h3] @ Wcat^T + b   (K=64 -> N=64)
// ---------------------------------------------------------------------------
__global__ __launch_bounds__(128) void sage_in_kernel(
    const float* __restrict__ h3, const float* __restrict__ agg1,
    const float* __restrict__ cnt, const float* __restrict__ Wl,
    const float* __restrict__ Wr, const float* __restrict__ b,
    float* __restrict__ h4, int nN) {
  __shared__ _Float16 sW[64 * 64];
  __shared__ float sBias[64];
  __shared__ _Float16 sA[4][16 * 64];
  int tid = threadIdx.x, wave = tid >> 5, lane = tid & 31;
  for (int t = tid; t < 64 * 64; t += 128) {
    int o = t >> 6, k = t & 63;
    float v = (k < 32) ? Wl[o * 32 + k] : Wr[o * 32 + (k - 32)];
    sW[t] = (_Float16)v;
  }
  for (int o = tid; o < 64; o += 128) sBias[o] = b[o];
  __syncthreads();

  int base = blockIdx.x * 64 + wave * 16;
  for (int t = lane; t < 16 * 64; t += 32) {
    int r = t >> 6, k = t & 63;
    int gm = base + r;
    int gmc = (gm < nN) ? gm : (nN - 1);
    float v;
    if (k < 32) v = agg1[(long long)gmc * 32 + k] / fmaxf(cnt[gmc], 1.0f);
    else        v = h3[(long long)gmc * 32 + (k - 32)];
    sA[wave][t] = (_Float16)((gm < nN) ? v : 0.0f);
  }
  __syncthreads();
  int rows = nN - base; rows = rows < 0 ? 0 : (rows > 16 ? 16 : rows);
  wave_gemm_global<64, 64>(sA[wave], sW, sBias,
                           h4 + (long long)base * 64, 64, rows, 64, lane, false);
}

// ---------------------------------------------------------------------------
// Kernel 3: sage_out (K=128 -> 32) fused with full decoder (32->64->64->10)
// ---------------------------------------------------------------------------
__global__ __launch_bounds__(128) void sage_out_decoder_kernel(
    const float* __restrict__ h4, const float* __restrict__ agg2,
    const float* __restrict__ cnt, const float* __restrict__ Wl,
    const float* __restrict__ Wr, const float* __restrict__ b,
    LBSD d0, LBSD d1, LBSD d2, float* __restrict__ out, int nN) {
  __shared__ _Float16 sWso[32 * 128];
  __shared__ float sBso[32];
  __shared__ _Float16 sW0[64 * 32];
  __shared__ _Float16 sW1[64 * 64];
  __shared__ _Float16 sW2[16 * 64];   // d2 padded 10 -> 16 rows
  __shared__ float sB0[64], sB1[64], sB2[16];
  __shared__ _Float16 sA[4][16 * 128];
  __shared__ _Float16 sAc[4][16 * 64];

  int tid = threadIdx.x, wave = tid >> 5, lane = tid & 31;
  for (int t = tid; t < 32 * 128; t += 128) {
    int o = t >> 7, k = t & 127;
    float v = (k < 64) ? Wl[o * 64 + k] : Wr[o * 64 + (k - 64)];
    sWso[t] = (_Float16)v;
  }
  for (int o = tid; o < 32; o += 128) sBso[o] = b[o];
  stage_lbsd(d0, sW0, sB0, 64, 32, tid, 128);
  stage_lbsd(d1, sW1, sB1, 64, 64, tid, 128);
  for (int t = tid; t < 16 * 64; t += 128) {   // d2 padded
    int o = t >> 6, k = t & 63;
    float v = 0.0f;
    if (o < 10) { float s = d2.g[o] * rsqrtf(d2.var[o] + EPS); v = d2.W[o * 64 + k] * s; }
    sW2[t] = (_Float16)v;
  }
  for (int o = tid; o < 16; o += 128) {
    float v = 0.0f;
    if (o < 10) { float s = d2.g[o] * rsqrtf(d2.var[o] + EPS); v = (d2.b[o] - d2.mu[o]) * s + d2.be[o]; }
    sB2[o] = v;
  }
  __syncthreads();

  int base = blockIdx.x * 64 + wave * 16;
  for (int t = lane; t < 16 * 128; t += 32) {
    int r = t >> 7, k = t & 127;
    int gm = base + r;
    int gmc = (gm < nN) ? gm : (nN - 1);
    float v;
    if (k < 64) v = agg2[(long long)gmc * 64 + k] / fmaxf(cnt[gmc], 1.0f);
    else        v = h4[(long long)gmc * 64 + (k - 64)];
    sA[wave][t] = (_Float16)((gm < nN) ? v : 0.0f);
  }
  __syncthreads();

  wave_gemm_lds<128, 32>(sA[wave], sWso, sBso, sAc[wave], lane, false); // sage_out
  __syncthreads();
  wave_gemm_lds<32, 64>(sAc[wave], sW0, sB0, sA[wave], lane, true);     // d0
  __syncthreads();
  wave_gemm_lds<64, 64>(sA[wave], sW1, sB1, sAc[wave], lane, true);     // d1
  __syncthreads();
  int rows = nN - base; rows = rows < 0 ? 0 : (rows > 16 ? 16 : rows);
  wave_gemm_global<64, 16>(sAc[wave], sW2, sB2,
                           out + (long long)base * 10, 10, rows, 10, lane, true);
}

// ---------------------------------------------------------------------------
extern "C" void kernel_launch(void* const* d_in, const int* in_sizes, int n_in,
                              void* d_out, int out_size, void* d_ws, size_t ws_size,
                              hipStream_t stream) {
  const float* x = (const float*)d_in[0];
  const int* ei = (const int*)d_in[1];
  int nN = in_sizes[0] / 128;
  int nE = in_sizes[1] / 2;

  auto L = [&](int base) {
    return LBSD{(const float*)d_in[base + 0], (const float*)d_in[base + 1],
                (const float*)d_in[base + 2], (const float*)d_in[base + 3],
                (const float*)d_in[base + 4], (const float*)d_in[base + 5]};
  };
  // setup_inputs() dict order: x, edge_index, e0(2), e1(8), e2(14),
  // d0(20), d1(26), d2(32), si_Wl(38) si_Wr si_b, so_Wl(41) so_Wr so_b
  LBSD e0 = L(2), e1 = L(8), e2 = L(14);
  LBSD d0 = L(20), d1 = L(26), d2 = L(32);
  const float* si_Wl = (const float*)d_in[38];
  const float* si_Wr = (const float*)d_in[39];
  const float* si_b  = (const float*)d_in[40];
  const float* so_Wl = (const float*)d_in[41];
  const float* so_Wr = (const float*)d_in[42];
  const float* so_b  = (const float*)d_in[43];

  float* ws   = (float*)d_ws;
  float* h3   = ws;                         // N*32
  float* h4   = h3 + (size_t)nN * 32;       // N*64
  float* agg1 = h4 + (size_t)nN * 64;       // N*32
  float* agg2 = agg1 + (size_t)nN * 32;     // N*64
  float* cnt  = agg2 + (size_t)nN * 64;     // N
  const int* src = ei;
  const int* dst = ei + nE;

  // agg1|agg2|cnt are contiguous: zero in one pass (graph-replay safe)
  zero_kernel<<<2048, 256, 0, stream>>>(agg1, (size_t)nN * 97);

  int nodeBlocks = (nN + 63) / 64;
  encoder_kernel<<<nodeBlocks, 128, 0, stream>>>(x, e0, e1, e2, h3, nN);
  degree_kernel<<<(nE + 255) / 256, 256, 0, stream>>>(dst, cnt, nE);

  long long t32 = (long long)nE * 32;
  scatter_add<32><<<(unsigned)((t32 + 255) / 256), 256, 0, stream>>>(src, dst, h3, agg1, nE);
  sage_in_kernel<<<nodeBlocks, 128, 0, stream>>>(h3, agg1, cnt, si_Wl, si_Wr, si_b, h4, nN);

  long long t64 = (long long)nE * 64;
  scatter_add<64><<<(unsigned)((t64 + 255) / 256), 256, 0, stream>>>(src, dst, h4, agg2, nE);
  sage_out_decoder_kernel<<<nodeBlocks, 128, 0, stream>>>(
      h4, agg2, cnt, so_Wl, so_Wr, so_b, d0, d1, d2, (float*)d_out, nN);
}
